// SoftAgg_61065845014739
// MI455X (gfx1250) — matrix operand order repaired and checked by reading, not compiled, not run on hardware
//
#include <hip/hip_runtime.h>
#include <hip/hip_bf16.h>

typedef __attribute__((ext_vector_type(16))) __bf16 v16bf;
typedef __attribute__((ext_vector_type(8)))  __bf16 v8bf;
typedef __attribute__((ext_vector_type(8)))  float  v8f;
typedef __attribute__((ext_vector_type(4)))  float  v4f;

#define B_DIM 2
#define E_DIM 32768
#define D_DIM 512
#define G_DIM 2048

#define WMMA_BF16(A, Bm, C) \
  __builtin_amdgcn_wmma_f32_16x16x32_bf16(false, (A), false, (Bm), (short)0, (C), false, false)

__device__ __forceinline__ v16bf cat16(v8bf a, v8bf b) {
  return __builtin_shufflevector(a, b, 0, 1, 2, 3, 4, 5, 6, 7, 8, 9, 10, 11, 12, 13, 14, 15);
}

// A fragment (16x32 MxK): lane<16 holds K ka..ka+7 and ka+16..ka+23 (ka=k0), lane>=16: +8.
__device__ __forceinline__ v16bf load_a(const __bf16* __restrict__ row, int ka) {
  return cat16(*(const v8bf*)(row + ka), *(const v8bf*)(row + ka + 16));
}
// B fragment (32x16 KxN): lane<16 holds K kb..kb+7 and kb+16..kb+23 (kb=k0), lane>=16: +8.
__device__ __forceinline__ v16bf load_b(const __bf16* __restrict__ row, int kb) {
  return cat16(*(const v8bf*)(row + kb), *(const v8bf*)(row + kb + 16));
}

// Split fp32 -> bf16 hi + bf16 lo (residual), 8 elements per thread.
__global__ __launch_bounds__(256) void ksplit(const float* __restrict__ src,
                                              __bf16* __restrict__ hi,
                                              __bf16* __restrict__ lo, long long n8) {
  const size_t t = (size_t)blockIdx.x * blockDim.x + threadIdx.x;
  if ((long long)t >= n8) return;
  const size_t base = t * 8;
  const v4f p = *(const v4f*)(src + base);
  const v4f q = *(const v4f*)(src + base + 4);
  v8bf h, l;
#pragma unroll
  for (int i = 0; i < 4; ++i) {
    __bf16 h0 = (__bf16)p[i]; h[i] = h0;     l[i] = (__bf16)(p[i] - (float)h0);
    __bf16 h1 = (__bf16)q[i]; h[4 + i] = h1; l[4 + i] = (__bf16)(q[i] - (float)h1);
  }
  *(v8bf*)(hi + base) = h;
  *(v8bf*)(lo + base) = l;
}

// K0: init group tables.
__global__ __launch_bounds__(256) void k0_init(float* __restrict__ mmax,
                                               float* __restrict__ denom,
                                               float* __restrict__ num) {
  size_t i = (size_t)blockIdx.x * blockDim.x + threadIdx.x;
  const size_t n = (size_t)G_DIM * B_DIM * D_DIM;
  if (i < n) { mmax[i] = -__builtin_inff(); denom[i] = 0.f; num[i] = 0.f; }
}

// K1: lt = x@Wg^T + bg ; vt = x@Wf^T + bf ; inline segment-max into mmax.
// Each wave: 64 M-rows x 16 N-cols (4 mtiles share register-resident B fragments).
// Grid: 1024 m-groups * 4 blocks; 8 waves/block cover 32 N-tiles.
__global__ __launch_bounds__(256) void k1_gemm_fg(
    const __bf16* __restrict__ xh, const __bf16* __restrict__ xl,
    const int* __restrict__ ix,
    const __bf16* __restrict__ wfh, const __bf16* __restrict__ wfl,
    const float* __restrict__ bfv,
    const __bf16* __restrict__ wgh, const __bf16* __restrict__ wgl,
    const float* __restrict__ bgv,
    float* __restrict__ lt, float* __restrict__ vt, float* __restrict__ mmax) {
  const int lane = threadIdx.x & 31;
  const int wave = threadIdx.x >> 5;
  const int mg = blockIdx.x >> 2;                  // 64-row m-group
  const int ntile = (blockIdx.x & 3) * 8 + wave;
  const int m0 = mg * 64;
  const int n0 = ntile * 16;
  const int col = lane & 15;
  const int h16 = lane >> 4;

  const __bf16* wg_h = wgh + (size_t)(n0 + col) * D_DIM;
  const __bf16* wg_l = wgl + (size_t)(n0 + col) * D_DIM;
  const __bf16* wf_h = wfh + (size_t)(n0 + col) * D_DIM;
  const __bf16* wf_l = wfl + (size_t)(n0 + col) * D_DIM;
  const __bf16* x_h  = xh  + (size_t)(m0 + col) * D_DIM;
  const __bf16* x_l  = xl  + (size_t)(m0 + col) * D_DIM;

  v8f cg[4] = {{}, {}, {}, {}};
  v8f cf[4] = {{}, {}, {}, {}};
  for (int k0 = 0; k0 < D_DIM; k0 += 32) {
    const int kb = k0 + h16 * 8;
    const v16bf gh = load_b(wg_h, kb);
    const v16bf gl = load_b(wg_l, kb);
    const v16bf fh = load_b(wf_h, kb);
    const v16bf fl = load_b(wf_l, kb);
    const int ka = k0 + h16 * 8;
#pragma unroll
    for (int mt = 0; mt < 4; ++mt) {
      const v16bf ah = load_a(x_h + (size_t)mt * 16 * D_DIM, ka);
      const v16bf al = load_a(x_l + (size_t)mt * 16 * D_DIM, ka);
      // bf16x3 split product: near-fp32 accuracy, f32 accumulate.
      cg[mt] = WMMA_BF16(ah, gh, cg[mt]);
      cg[mt] = WMMA_BF16(ah, gl, cg[mt]);
      cg[mt] = WMMA_BF16(al, gh, cg[mt]);
      cf[mt] = WMMA_BF16(ah, fh, cf[mt]);
      cf[mt] = WMMA_BF16(ah, fl, cf[mt]);
      cf[mt] = WMMA_BF16(al, fh, cf[mt]);
    }
  }
  const float bg_s = bgv[n0 + col];
  const float bf_s = bfv[n0 + col];
  const int ncol = n0 + col;
#pragma unroll
  for (int mt = 0; mt < 4; ++mt) {
#pragma unroll
    for (int v = 0; v < 8; ++v) {
      const int r = m0 + mt * 16 + v + h16 * 8;   // output row (b*E + e)
      const int b = r >> 15;                       // E = 2^15
      const int e = r & (E_DIM - 1);
      const int g = ix[e];
      const float lg = cg[mt][v] + bg_s;
      const float lf = cf[mt][v] + bf_s;
      lt[(size_t)r * D_DIM + ncol] = lg;
      vt[(size_t)r * D_DIM + ncol] = lf;
      __hip_atomic_fetch_max(mmax + ((size_t)g * B_DIM + b) * D_DIM + ncol, lg,
                             __ATOMIC_RELAXED, __HIP_MEMORY_SCOPE_AGENT);
    }
  }
}

// K2: e = exp(lt - m[g]); denom[g] += e; num[g] += vt * e.  (tables L2-resident)
__global__ __launch_bounds__(256) void k2_softnum(
    const int* __restrict__ ix, const float* __restrict__ lt,
    const float* __restrict__ vt, const float* __restrict__ mmax,
    float* __restrict__ denom, float* __restrict__ num) {
  const size_t i4 = (size_t)blockIdx.x * blockDim.x + threadIdx.x;  // one float4
  const int n4 = (int)(i4 & (D_DIM / 4 - 1));
  const size_t r = i4 >> 7;
  const int b = (int)(r >> 15);
  const int e = (int)(r & (E_DIM - 1));
  const int g = ix[e];
  const size_t src = r * D_DIM + (size_t)n4 * 4;
  const size_t dst = ((size_t)g * B_DIM + b) * D_DIM + (size_t)n4 * 4;
  const v4f l = *(const v4f*)(lt + src);
  const v4f v = *(const v4f*)(vt + src);
  const v4f mv = *(const v4f*)(mmax + dst);
#pragma unroll
  for (int j = 0; j < 4; ++j) {
    const float ex = __expf(l[j] - mv[j]);
    __hip_atomic_fetch_add(denom + dst + j, ex, __ATOMIC_RELAXED, __HIP_MEMORY_SCOPE_AGENT);
    __hip_atomic_fetch_add(num + dst + j, v[j] * ex, __ATOMIC_RELAXED, __HIP_MEMORY_SCOPE_AGENT);
  }
}

// K2b: y[b,g,:] = num[g,b,:] / denom[g,b,:], stored pre-split as bf16 hi/lo.
__global__ __launch_bounds__(256) void k2b_norm(const float* __restrict__ num,
                                                const float* __restrict__ denom,
                                                __bf16* __restrict__ yh,
                                                __bf16* __restrict__ yl) {
  size_t i = (size_t)blockIdx.x * blockDim.x + threadIdx.x;
  const size_t n = (size_t)G_DIM * B_DIM * D_DIM;
  if (i >= n) return;
  const int d = (int)(i & (D_DIM - 1));
  const size_t gb = i >> 9;
  const int b = (int)(gb & (B_DIM - 1));
  const size_t g = gb >> 1;
  const float v = num[i] / denom[i];
  const __bf16 h = (__bf16)v;
  const size_t o = ((size_t)b * G_DIM + g) * D_DIM + d;
  yh[o] = h;
  yl[o] = (__bf16)(v - (float)h);
}

// K3: outg = y @ Wh^T + bh  over (B*G) x D rows.
__global__ __launch_bounds__(256) void k3_gemm_h(
    const __bf16* __restrict__ yh, const __bf16* __restrict__ yl,
    const __bf16* __restrict__ whh, const __bf16* __restrict__ whl,
    const float* __restrict__ bhv, float* __restrict__ outg) {
  const int lane = threadIdx.x & 31;
  const int wave = threadIdx.x >> 5;
  const int mtile = blockIdx.x >> 2;
  const int ntile = (blockIdx.x & 3) * 8 + wave;
  const int m0 = mtile * 16;
  const int n0 = ntile * 16;
  const int col = lane & 15;
  const int h16 = lane >> 4;

  const __bf16* y_h = yh + (size_t)(m0 + col) * D_DIM;
  const __bf16* y_l = yl + (size_t)(m0 + col) * D_DIM;
  const __bf16* w_h = whh + (size_t)(n0 + col) * D_DIM;
  const __bf16* w_l = whl + (size_t)(n0 + col) * D_DIM;
  v8f c = {};
  for (int k0 = 0; k0 < D_DIM; k0 += 32) {
    const int kb = k0 + h16 * 8;
    const int ka = k0 + h16 * 8;
    const v16bf bh = load_b(w_h, kb);
    const v16bf bl = load_b(w_l, kb);
    const v16bf ah = load_a(y_h, ka);
    const v16bf al = load_a(y_l, ka);
    c = WMMA_BF16(ah, bh, c);
    c = WMMA_BF16(ah, bl, c);
    c = WMMA_BF16(al, bh, c);
  }
  const float bs = bhv[n0 + col];
#pragma unroll
  for (int v = 0; v < 8; ++v) {
    const int r = m0 + v + h16 * 8;
    outg[(size_t)r * D_DIM + n0 + col] = c[v] + bs;
  }
}

// K4: out[b,e,:] = outg[b, ix[e], :]   (source L2-resident; NT final stores)
__global__ __launch_bounds__(128) void k4_gather(const int* __restrict__ ix,
                                                 const float* __restrict__ outg,
                                                 float* __restrict__ out) {
  const int row = blockIdx.x;            // b*E + e
  const int b = row >> 15;
  const int e = row & (E_DIM - 1);
  const int g = ix[e];
  const v4f* src = (const v4f*)(outg + ((size_t)b * G_DIM + g) * D_DIM);
  v4f* dst = (v4f*)(out + (size_t)row * D_DIM);
  const v4f val = src[threadIdx.x];
  __builtin_nontemporal_store(val, dst + threadIdx.x);
}

extern "C" void kernel_launch(void* const* d_in, const int* in_sizes, int n_in,
                              void* d_out, int out_size, void* d_ws, size_t ws_size,
                              hipStream_t stream) {
  (void)in_sizes; (void)n_in; (void)out_size; (void)ws_size;
  const float* x  = (const float*)d_in[0];
  const int*   ix = (const int*)d_in[1];
  const float* Wf = (const float*)d_in[2];
  const float* bf = (const float*)d_in[3];
  const float* Wg = (const float*)d_in[4];
  const float* bg = (const float*)d_in[5];
  const float* Wh = (const float*)d_in[6];
  const float* bh = (const float*)d_in[7];
  float* out = (float*)d_out;

  const size_t SZ_EDGE = (size_t)B_DIM * E_DIM * D_DIM;  // 33.5M elems
  const size_t SZ_GRP  = (size_t)G_DIM * B_DIM * D_DIM;  // 2.1M elems
  const size_t SZ_W    = (size_t)D_DIM * D_DIM;          // 262144 elems

  float* lt    = (float*)d_ws;
  float* vt    = lt + SZ_EDGE;
  float* mmax  = vt + SZ_EDGE;
  float* denom = mmax + SZ_GRP;
  float* num   = denom + SZ_GRP;
  float* outg  = num + SZ_GRP;
  __bf16* xh   = (__bf16*)(outg + SZ_GRP);
  __bf16* xl   = xh + SZ_EDGE;
  __bf16* yh   = xl + SZ_EDGE;
  __bf16* yl   = yh + SZ_GRP;
  __bf16* wgh  = yl + SZ_GRP;
  __bf16* wgl  = wgh + SZ_W;
  __bf16* wfh  = wgl + SZ_W;
  __bf16* wfl  = wfh + SZ_W;
  __bf16* whh  = wfl + SZ_W;
  __bf16* whl  = whh + SZ_W;

  // Prolog: one-time fp32 -> bf16 hi/lo splits (hoists all conversion VALU).
  ksplit<<<(int)(SZ_EDGE / 8 / 256), 256, 0, stream>>>(x, xh, xl, (long long)(SZ_EDGE / 8));
  ksplit<<<(int)(SZ_W / 8 / 256), 256, 0, stream>>>(Wg, wgh, wgl, (long long)(SZ_W / 8));
  ksplit<<<(int)(SZ_W / 8 / 256), 256, 0, stream>>>(Wf, wfh, wfl, (long long)(SZ_W / 8));
  ksplit<<<(int)(SZ_W / 8 / 256), 256, 0, stream>>>(Wh, whh, whl, (long long)(SZ_W / 8));
  k0_init<<<(int)((SZ_GRP + 255) / 256), 256, 0, stream>>>(mmax, denom, num);

  k1_gemm_fg<<<(B_DIM * E_DIM / 64) * 4, 256, 0, stream>>>(
      xh, xl, ix, wfh, wfl, bf, wgh, wgl, bg, lt, vt, mmax);
  k2_softnum<<<(int)((B_DIM * (size_t)E_DIM * (D_DIM / 4)) / 256), 256, 0, stream>>>(
      ix, lt, vt, mmax, denom, num);
  k2b_norm<<<(int)((SZ_GRP + 255) / 256), 256, 0, stream>>>(num, denom, yh, yl);
  k3_gemm_h<<<(B_DIM * G_DIM / 16) * 4, 256, 0, stream>>>(yh, yl, whh, whl, bh, outg);
  k4_gather<<<B_DIM * E_DIM, 128, 0, stream>>>(ix, outg, out);
}
